// MultiHeadAttention_6476810682498
// MI455X (gfx1250) — matrix operand order repaired and checked by reading, not compiled, not run on hardware
//
#include <hip/hip_runtime.h>
#include <hip/hip_bf16.h>

// MI455X / gfx1250 multi-head causal attention, f16 WMMA end-to-end, f32 accum.
// B=4, S=2048, D=1024, H=16, HD=64.

typedef __attribute__((ext_vector_type(16))) _Float16 v16h;
typedef __attribute__((ext_vector_type(8)))  float    v8f;
typedef __attribute__((ext_vector_type(4)))  _Float16 v4h;

union FragH {            // one 16x32 A-fragment or 32x16 B-fragment (16 f16 / lane)
  v16h v;
  uint4 q[2];
};

static __device__ __forceinline__ v8f wmma_f16(const FragH& a, const FragH& b, v8f c) {
  return __builtin_amdgcn_wmma_f32_16x16x32_f16(
      /*neg_a=*/false, a.v, /*neg_b=*/false, b.v,
      /*c_mod=*/(short)0, c, /*reuse_a=*/false, /*reuse_b=*/false);
}

// ---------------------------------------------------------------------------
// fp32 -> f16 cast, 4 elements / thread
// ---------------------------------------------------------------------------
__global__ __launch_bounds__(256) void cast_f32_to_f16(
    const float* __restrict__ in, _Float16* __restrict__ out, int n4) {
  int i = blockIdx.x * blockDim.x + threadIdx.x;
  if (i >= n4) return;
  const float4 f = reinterpret_cast<const float4*>(in)[i];
  v4h r = { (_Float16)f.x, (_Float16)f.y, (_Float16)f.z, (_Float16)f.w };
  reinterpret_cast<v4h*>(out)[i] = r;
}

// ---------------------------------------------------------------------------
// GEMM: C[m,n] = sum_k A[m,k] * W[n,k] + bias[n]   (M=8192, N=K=1024)
// One wave computes a 16x64 tile. mode: 0=Q [B,H,S,64], 1=K [B,H,S,64],
// 2=V transposed [B,H,64,S], 3=final f32 out [M,1024].
// ---------------------------------------------------------------------------
__global__ __launch_bounds__(256) void gemm_xwT(
    const _Float16* __restrict__ A,
    const _Float16* __restrict__ W,
    const float*    __restrict__ bias,
    void* __restrict__ outp,
    int mode)
{
  const int lane  = threadIdx.x & 31;
  const int wid   = threadIdx.x >> 5;
  const int gw    = blockIdx.x * 8 + wid;      // 8192 waves total
  const int mbase = (gw >> 4) * 16;
  const int nbase = (gw & 15) * 64;
  const int ml    = lane & 15;
  const int half  = lane >> 4;

  v8f acc[4] = {};

  const _Float16* arow = A + (size_t)(mbase + ml) * 1024;
  for (int k0 = 0; k0 < 1024; k0 += 32) {
    FragH a;
    a.q[0] = *(const uint4*)(arow + k0 + half * 8);
    a.q[1] = *(const uint4*)(arow + k0 + 16 + half * 8);
    __builtin_prefetch(arow + k0 + 64, 0, 1);
#pragma unroll
    for (int nt = 0; nt < 4; ++nt) {
      const _Float16* wrow =
          W + (size_t)(nbase + nt * 16 + ml) * 1024 + k0 + half * 8;
      FragH b;
      b.q[0] = *(const uint4*)(wrow);
      b.q[1] = *(const uint4*)(wrow + 16);
      acc[nt] = wmma_f16(a, b, acc[nt]);
    }
  }

#pragma unroll
  for (int nt = 0; nt < 4; ++nt) {
    const int n  = nbase + nt * 16 + ml;
    const float bv = bias[n];
#pragma unroll
    for (int r = 0; r < 8; ++r) {
      const int Mrow = mbase + r + 8 * half;       // C layout: M = r + 8*(lane/16)
      const float val = acc[nt][r] + bv;
      if (mode == 3) {
        ((float*)outp)[(size_t)Mrow * 1024 + n] = val;
      } else {
        const int bb = Mrow >> 11, s = Mrow & 2047;
        const int h  = n >> 6,    hd = n & 63;
        _Float16* o = (_Float16*)outp;
        if (mode == 2)  // V transposed: [B,H,64,S]
          o[((size_t)(bb * 16 + h) * 64 + hd) * 2048 + s] = (_Float16)val;
        else            // Q/K: [B,H,S,64]
          o[((size_t)(bb * 16 + h) * 2048 + s) * 64 + hd] = (_Float16)val;
      }
    }
  }
}

// ---------------------------------------------------------------------------
// Flash attention: one wave per (b,h,16-row q tile), streaming 32-wide
// k-blocks with online softmax. Causal + pad mask.
// Q,K: [B*H, S, 64] f16.  Vt: [B*H, 64, S] f16.  ctx: [B*S, 1024] f16.
// ---------------------------------------------------------------------------
#define PLDS_STRIDE 48

__global__ __launch_bounds__(128) void flash_attn(
    const _Float16* __restrict__ Q,
    const _Float16* __restrict__ K,
    const _Float16* __restrict__ Vt,
    const int*      __restrict__ amask,   // [B,S]
    _Float16*       __restrict__ ctx)
{
  __shared__ _Float16 plds[4][16 * PLDS_STRIDE];

  const int lane  = threadIdx.x & 31;
  const int wid   = threadIdx.x >> 5;
  const int qt    = blockIdx.x * 4 + wid;   // 8192 q-tiles
  const int bh    = qt >> 7;                // (b*16+h)
  const int qbase = (qt & 127) << 4;
  const int b     = bh >> 4;
  const int h     = bh & 15;
  const int ml    = lane & 15;
  const int half  = lane >> 4;

  // Q fragments for d=0..31 and d=32..63, loaded once.
  const _Float16* qrow = Q + ((size_t)bh * 2048 + qbase + ml) * 64;
  FragH aqlo, aqhi;
  aqlo.q[0] = *(const uint4*)(qrow + half * 8);
  aqlo.q[1] = *(const uint4*)(qrow + 16 + half * 8);
  aqhi.q[0] = *(const uint4*)(qrow + 32 + half * 8);
  aqhi.q[1] = *(const uint4*)(qrow + 48 + half * 8);

  float mrow[8], lrow[8];
  v8f oacc[4] = {};
#pragma unroll
  for (int r = 0; r < 8; ++r) { mrow[r] = -3.0e38f; lrow[r] = 0.0f; }

  const int nblocks = (qbase + 16 + 31) >> 5;   // causal upper bound
  _Float16* pw = &plds[wid][0];

  for (int kb = 0; kb < nblocks; ++kb) {
    const int kc0 = kb << 5;

    // ---- scores: two 16x16 tiles of Q*K^T over d=64 (2 WMMAs each) ----
    v8f s0 = {}, s1 = {};
    {
      const _Float16* kr0 = K + ((size_t)bh * 2048 + kc0 + ml) * 64 + half * 8;
      FragH bk;
      bk.q[0] = *(const uint4*)(kr0);
      bk.q[1] = *(const uint4*)(kr0 + 16);
      s0 = wmma_f16(aqlo, bk, s0);
      bk.q[0] = *(const uint4*)(kr0 + 32);
      bk.q[1] = *(const uint4*)(kr0 + 48);
      s0 = wmma_f16(aqhi, bk, s0);

      const _Float16* kr1 = kr0 + 16 * 64;
      bk.q[0] = *(const uint4*)(kr1);
      bk.q[1] = *(const uint4*)(kr1 + 16);
      s1 = wmma_f16(aqlo, bk, s1);
      bk.q[0] = *(const uint4*)(kr1 + 32);
      bk.q[1] = *(const uint4*)(kr1 + 48);
      s1 = wmma_f16(aqhi, bk, s1);
    }

    const int col0 = kc0 + ml;
    const int col1 = col0 + 16;
    const float pb0 = (amask[b * 2048 + col0] == 0) ? -1.0e9f : 0.0f;
    const float pb1 = (amask[b * 2048 + col1] == 0) ? -1.0e9f : 0.0f;

    // ---- online softmax per row (row = 16 lanes of one half-wave) ----
#pragma unroll
    for (int r = 0; r < 8; ++r) {
      const int qr = qbase + r + 8 * half;
      float v0 = s0[r] * 0.125f + pb0;   // scale = 64^-0.5
      float v1 = s1[r] * 0.125f + pb1;
      if (col0 > qr) v0 = -3.0e38f;      // causal
      if (col1 > qr) v1 = -3.0e38f;

      float mx = fmaxf(v0, v1);
#pragma unroll
      for (int off = 1; off < 16; off <<= 1)
        mx = fmaxf(mx, __shfl_xor(mx, off, 32));
      const float mnew  = fmaxf(mrow[r], mx);
      const float alpha = __expf(mrow[r] - mnew);
      const float p0 = __expf(v0 - mnew);
      const float p1 = __expf(v1 - mnew);
      float ps = p0 + p1;
#pragma unroll
      for (int off = 1; off < 16; off <<= 1)
        ps += __shfl_xor(ps, off, 32);
      lrow[r] = lrow[r] * alpha + ps;
      mrow[r] = mnew;
      oacc[0][r] *= alpha; oacc[1][r] *= alpha;
      oacc[2][r] *= alpha; oacc[3][r] *= alpha;

      // stash P tile (C layout) to LDS as f16
      const int m = r + 8 * half;
      pw[m * PLDS_STRIDE + ml]      = (_Float16)p0;
      pw[m * PLDS_STRIDE + 16 + ml] = (_Float16)p1;
    }

    asm volatile("s_wait_dscnt 0" ::: "memory");   // LDS RAW fence

    // reload P in A-fragment layout (16x32, K = 32 seq columns)
    FragH ap;
    ap.q[0] = *(const uint4*)(pw + ml * PLDS_STRIDE + half * 8);
    ap.q[1] = *(const uint4*)(pw + ml * PLDS_STRIDE + 16 + half * 8);

    // ---- O += P @ V_block, V transposed so B-frag rows are contiguous ----
#pragma unroll
    for (int dt = 0; dt < 4; ++dt) {
      const _Float16* vr =
          Vt + ((size_t)bh * 64 + dt * 16 + ml) * 2048 + kc0 + half * 8;
      FragH bv;
      bv.q[0] = *(const uint4*)(vr);
      bv.q[1] = *(const uint4*)(vr + 16);
      oacc[dt] = wmma_f16(ap, bv, oacc[dt]);
    }
  }

  // ---- normalize and write context: ctx[b, s, h*64 + d] ----
#pragma unroll
  for (int r = 0; r < 8; ++r) {
    const float inv = 1.0f / fmaxf(lrow[r], 1e-20f);
    const int s = qbase + r + 8 * half;
    _Float16* orow = ctx + ((size_t)(b * 2048 + s)) * 1024 + h * 64;
#pragma unroll
    for (int dt = 0; dt < 4; ++dt)
      orow[dt * 16 + ml] = (_Float16)(oacc[dt][r] * inv);
  }
}

// ---------------------------------------------------------------------------
extern "C" void kernel_launch(void* const* d_in, const int* in_sizes, int n_in,
                              void* d_out, int out_size, void* d_ws, size_t ws_size,
                              hipStream_t stream) {
  (void)in_sizes; (void)n_in; (void)out_size; (void)ws_size;

  const float* x  = (const float*)d_in[0];
  const int*   am = (const int*)  d_in[1];
  const float* Wq = (const float*)d_in[2];
  const float* bq = (const float*)d_in[3];
  const float* Wk = (const float*)d_in[4];
  const float* bk = (const float*)d_in[5];
  const float* Wv = (const float*)d_in[6];
  const float* bv = (const float*)d_in[7];
  const float* Wo = (const float*)d_in[8];
  const float* bo = (const float*)d_in[9];

  const size_t NX = 8388608;   // B*S*D
  const size_t NW = 1048576;   // D*D

  _Float16* xh  = (_Float16*)d_ws;
  _Float16* wqh = xh  + NX;
  _Float16* wkh = wqh + NW;
  _Float16* wvh = wkh + NW;
  _Float16* woh = wvh + NW;
  _Float16* qh  = woh + NW;
  _Float16* kh  = qh  + NX;
  _Float16* vth = kh  + NX;
  _Float16* ch  = vth + NX;    // total ws use: 5*16MB + 4*2MB = 88 MB

  cast_f32_to_f16<<<8192, 256, 0, stream>>>(x,  xh,  (int)(NX / 4));
  cast_f32_to_f16<<<1024, 256, 0, stream>>>(Wq, wqh, (int)(NW / 4));
  cast_f32_to_f16<<<1024, 256, 0, stream>>>(Wk, wkh, (int)(NW / 4));
  cast_f32_to_f16<<<1024, 256, 0, stream>>>(Wv, wvh, (int)(NW / 4));
  cast_f32_to_f16<<<1024, 256, 0, stream>>>(Wo, woh, (int)(NW / 4));

  gemm_xwT<<<1024, 256, 0, stream>>>(xh, wqh, bq, qh,  0);  // Q
  gemm_xwT<<<1024, 256, 0, stream>>>(xh, wkh, bk, kh,  1);  // K
  gemm_xwT<<<1024, 256, 0, stream>>>(xh, wvh, bv, vth, 2);  // V (transposed)

  flash_attn<<<2048, 128, 0, stream>>>(qh, kh, vth, am, ch);

  gemm_xwT<<<1024, 256, 0, stream>>>(ch, woh, bo, d_out, 3);  // out proj, f32
}
